// Projector_69664369541338
// MI455X (gfx1250) — matrix operand order repaired and checked by reading, not compiled, not run on hardware
//
#include <hip/hip_runtime.h>

#define N_B  16
#define HW_  3136
#define LQ   64
#define CIN  256
#define COUT 512
#define DD   256

#if defined(__has_builtin)
#if __has_builtin(__builtin_amdgcn_global_load_async_to_lds_b128)
#define HAVE_ASYNC_LDS 1
#endif
#endif

typedef __attribute__((ext_vector_type(16))) _Float16 v16h;
typedef __attribute__((ext_vector_type(8)))  float    v8f;

#if HAVE_ASYNC_LDS
// exact parameter types per hipcc diagnostic: int __vector_size__(16), AS1 src / AS3 dst
typedef int async_i4 __attribute__((vector_size(16)));
typedef __attribute__((address_space(1))) async_i4* g4p;
typedef __attribute__((address_space(3))) async_i4* l4p;
#endif

union FragH { v16h v; uint4 q[2]; };
union Pack4 { _Float16 h[4]; uint2 u; };

// ---- WMMA fragment loaders (row-major f16 source, ld multiple of 8 halves) ----
// A (16x32, MxK): lane L holds row L%16; halves 0-7 at K = kBase + (L/16)*8,
//                 halves 8-15 at +16.  Two 16B contiguous segments per lane.
__device__ __forceinline__ v16h frag_a(const _Float16* m, int ldh, int rowBase, int kBase) {
    int lane = threadIdx.x & 31;
    const _Float16* p = m + (size_t)(rowBase + (lane & 15)) * ldh + kBase + ((lane >> 4) << 3);
    FragH f;
    f.q[0] = *(const uint4*)p;
    f.q[1] = *(const uint4*)(p + 16);
    return f.v;
}
// B (32x16, KxN): lane L holds column L%16; halves 0-7 at K = kBase + (L/16)*8,
// halves 8-15 at +16.  Two 16B contiguous segments per lane.
__device__ __forceinline__ v16h frag_b(const _Float16* m, int ldh, int colBase, int kBase) {
    int lane = threadIdx.x & 31;
    const _Float16* p = m + (size_t)(colBase + (lane & 15)) * ldh + kBase + ((lane >> 4) << 3);
    FragH f;
    f.q[0] = *(const uint4*)p;
    f.q[1] = *(const uint4*)(p + 16);
    return f.v;
}

__device__ __forceinline__ v8f wmma_f16(v16h a, v16h b, v8f c) {
    return __builtin_amdgcn_wmma_f32_16x16x32_f16(false, a, false, b, (short)0, c, false, false);
}

// ---- tiny weight conversion kernels ----
__global__ void cvt_f32_to_f16(const float* __restrict__ in, _Float16* __restrict__ out, int n) {
    int i4 = (blockIdx.x * blockDim.x + threadIdx.x) * 4;
    if (i4 < n) {
        float4 v = *(const float4*)(in + i4);
        Pack4 pk;
        pk.h[0] = (_Float16)v.x; pk.h[1] = (_Float16)v.y;
        pk.h[2] = (_Float16)v.z; pk.h[3] = (_Float16)v.w;
        *(uint2*)(out + i4) = pk.u;
    }
}

// out[c][d] = in[d][c]  (in: rows x cols f32 -> out: cols x rows f16)
__global__ void cvt_transpose_f16(const float* __restrict__ in, _Float16* __restrict__ out,
                                  int rows, int cols) {
    int i = blockIdx.x * blockDim.x + threadIdx.x;
    if (i < rows * cols) {
        int c = i / rows, d = i % rows;
        out[(size_t)c * rows + d] = (_Float16)in[(size_t)d * cols + c];
    }
}

// ---- per-batch precompute:  U = (t*W2^T)*W1  (64x256),  t3T = (t*W3^T + b3)^T (512x64) ----
__global__ __launch_bounds__(256) void prep_kernel(
    const float* __restrict__ t, const float* __restrict__ b3,
    const _Float16* __restrict__ W1t, const _Float16* __restrict__ W2h,
    const _Float16* __restrict__ W3h,
    _Float16* __restrict__ U, _Float16* __restrict__ T3T) {
    __shared__ _Float16 Th [LQ][DD + 8];
    __shared__ _Float16 TQh[LQ][DD + 8];
    const int n = blockIdx.x;
    const int tid = threadIdx.x, wave = tid >> 5, lane = tid & 31;

    // stage t[n] (64x256 f32) -> f16 LDS
    const float* tp = t + (size_t)n * LQ * DD;
    for (int i = 0; i < 16; ++i) {
        int idx = tid + 256 * i;
        int row = idx >> 6, c4 = (idx & 63) << 2;
        float4 v = *(const float4*)(tp + (size_t)row * DD + c4);
        Pack4 pk;
        pk.h[0] = (_Float16)v.x; pk.h[1] = (_Float16)v.y;
        pk.h[2] = (_Float16)v.z; pk.h[3] = (_Float16)v.w;
        *(uint2*)&Th[row][c4] = pk.u;
    }
    __syncthreads();

    // t_q = Th * W2^T  -> TQh (64x256 f16).  64 tiles, 8 per wave, K=256.
    for (int i = 0; i < 8; ++i) {
        int id = wave + 8 * i;
        int tm = id >> 4, te = id & 15;
        v8f acc = {};
        for (int k = 0; k < 8; ++k)
            acc = wmma_f16(frag_a(&Th[0][0], DD + 8, tm * 16, k * 32),
                           frag_b(W2h, DD, te * 16, k * 32), acc);
        int col = te * 16 + (lane & 15);
        int rb  = tm * 16 + ((lane >> 4) << 3);
        for (int f = 0; f < 8; ++f) TQh[rb + f][col] = (_Float16)acc[f];
    }
    __syncthreads();

    // U = TQh * W1  (B from W1t[c][d], contiguous in d) -> global (64x256 f16)
    _Float16* Un = U + (size_t)n * LQ * CIN;
    for (int i = 0; i < 8; ++i) {
        int id = wave + 8 * i;
        int tm = id >> 4, tc = id & 15;
        v8f acc = {};
        for (int k = 0; k < 8; ++k)
            acc = wmma_f16(frag_a(&TQh[0][0], DD + 8, tm * 16, k * 32),
                           frag_b(W1t, DD, tc * 16, k * 32), acc);
        int col = tc * 16 + (lane & 15);
        int rb  = tm * 16 + ((lane >> 4) << 3);
        for (int f = 0; f < 8; ++f) Un[(size_t)(rb + f) * CIN + col] = (_Float16)acc[f];
    }

    // t3T[o][l] = sum_d W3[o][d]*t[l][d] + b3[o]   (512x64, stored o-major)
    _Float16* T3n = T3T + (size_t)n * COUT * LQ;
    for (int i = 0; i < 16; ++i) {
        int id = wave + 8 * i;         // 128 tiles: 32 o-tiles x 4 l-tiles
        int to = id >> 2, tl = id & 3;
        v8f acc = {};
        for (int k = 0; k < 8; ++k)
            acc = wmma_f16(frag_a(W3h, DD, to * 16, k * 32),
                           frag_b(&Th[0][0], DD + 8, tl * 16, k * 32), acc);
        int col = tl * 16 + (lane & 15);
        int rb  = to * 16 + ((lane >> 4) << 3);
        for (int f = 0; f < 8; ++f)
            T3n[(size_t)(rb + f) * LQ + col] = (_Float16)(acc[f] + b3[rb + f]);
    }
}

// ---- fused main kernel: per (batch, 64-row x-tile) ----
// S = Xh*U^T (64x64) -> softmax -> out = P*t3 + Xh*Wd^T + bd
__global__ __launch_bounds__(256) void main_kernel(
    const float* __restrict__ x, const _Float16* __restrict__ U,
    const _Float16* __restrict__ T3T, const _Float16* __restrict__ Wdh,
    const float* __restrict__ bd, float* __restrict__ out) {
    __shared__ _Float16 Xh [64][CIN + 8];  // 33.8 KB
    __shared__ _Float16 Ush[LQ][CIN + 8];  // 33.8 KB  (U staged via async DMA)
    __shared__ float    Sm [64][LQ + 1];   // 16.6 KB
    __shared__ _Float16 Pm [64][LQ + 8];   //  9.2 KB
    const int n = blockIdx.y;
    const int rowBase = blockIdx.x * 64;
    const int tid = threadIdx.x, wave = tid >> 5, lane = tid & 31;

    const _Float16* Un  = U   + (size_t)n * LQ * CIN;
    const _Float16* T3n = T3T + (size_t)n * COUT * LQ;

    // warm GL2 for the panels consumed in the output phase (global_prefetch_b8)
    __builtin_prefetch(T3n + (size_t)tid * 128, 0, 1);
    __builtin_prefetch(Wdh + (size_t)tid * 512, 0, 1);

    // ---- kick off async copy of U[n] (64x256 f16) into padded LDS ----
    // 2048 x 16B chunks; row stride in LDS = 528B (264 halves).
    {
        const char* usrc = (const char*)Un;
        char*       udst = (char*)&Ush[0][0];
        for (int i = 0; i < 8; ++i) {
            int c = tid + 256 * i;
            int row = c >> 5, off = (c & 31) * 16;
#if HAVE_ASYNC_LDS
            __builtin_amdgcn_global_load_async_to_lds_b128(
                (g4p)(usrc + (size_t)c * 16),
                (l4p)(udst + row * 528 + off),
                0, 0);
#else
            *(uint4*)(udst + row * 528 + off) = *(const uint4*)(usrc + (size_t)c * 16);
#endif
        }
    }

    // ---- stage x tile (64x256 f32, read ONCE) -> f16 LDS, overlapping the DMA ----
    const float* xp = x + ((size_t)n * HW_ + rowBase) * CIN;
    for (int i = 0; i < 16; ++i) {
        int idx = tid + 256 * i;
        int row = idx >> 6, c4 = (idx & 63) << 2;
        float4 v = *(const float4*)(xp + (size_t)row * CIN + c4);
        Pack4 pk;
        pk.h[0] = (_Float16)v.x; pk.h[1] = (_Float16)v.y;
        pk.h[2] = (_Float16)v.z; pk.h[3] = (_Float16)v.w;
        *(uint2*)&Xh[row][c4] = pk.u;
    }

#if HAVE_ASYNC_LDS
#if defined(__has_builtin) && __has_builtin(__builtin_amdgcn_s_wait_asynccnt)
    __builtin_amdgcn_s_wait_asynccnt(0);
#else
    asm volatile("s_wait_asynccnt 0x0" ::: "memory");
#endif
#endif
    __syncthreads();

    // logits S = Xh * U^T : 16 tiles (4x4), 2 per wave, K=256 (both operands in LDS)
    for (int i = 0; i < 2; ++i) {
        int id = wave + 8 * i;
        int tm = id >> 2, tl = id & 3;
        v8f acc = {};
        for (int k = 0; k < 8; ++k)
            acc = wmma_f16(frag_a(&Xh[0][0], CIN + 8, tm * 16, k * 32),
                           frag_b(&Ush[0][0], CIN + 8, tl * 16, k * 32), acc);
        int col = tl * 16 + (lane & 15);
        int rb  = tm * 16 + ((lane >> 4) << 3);
        for (int f = 0; f < 8; ++f) Sm[rb + f][col] = acc[f];
    }
    __syncthreads();

    // row softmax over L=64 (one thread per row)
    if (tid < 64) {
        float mx = -1e30f;
        for (int l = 0; l < LQ; ++l) mx = fmaxf(mx, Sm[tid][l]);
        float s = 0.f;
        for (int l = 0; l < LQ; ++l) { float e = __expf(Sm[tid][l] - mx); Sm[tid][l] = e; s += e; }
        float inv = 1.0f / s;
        for (int l = 0; l < LQ; ++l) Pm[tid][l] = (_Float16)(Sm[tid][l] * inv);
    }
    __syncthreads();

    // out = P * t3 (K=64)  +  Xh * Wd^T (K=256)  + bd
    float* op = out + ((size_t)n * HW_ + rowBase) * COUT;
    for (int i = 0; i < 16; ++i) {
        int id = wave + 8 * i;          // 128 tiles: 4 m-tiles x 32 o-tiles
        int tm = id >> 5, to = id & 31;
        v8f acc = {};
        for (int k = 0; k < 2; ++k)
            acc = wmma_f16(frag_a(&Pm[0][0], LQ + 8, tm * 16, k * 32),
                           frag_b(T3n, LQ, to * 16, k * 32), acc);
        for (int k = 0; k < 8; ++k)
            acc = wmma_f16(frag_a(&Xh[0][0], CIN + 8, tm * 16, k * 32),
                           frag_b(Wdh, CIN, to * 16, k * 32), acc);
        int col = to * 16 + (lane & 15);
        float bdv = bd[col];
        int rb = tm * 16 + ((lane >> 4) << 3);
        for (int f = 0; f < 8; ++f)
            op[(size_t)(rb + f) * COUT + col] = acc[f] + bdv;
    }
}

extern "C" void kernel_launch(void* const* d_in, const int* in_sizes, int n_in,
                              void* d_out, int out_size, void* d_ws, size_t ws_size,
                              hipStream_t stream) {
    const float* x  = (const float*)d_in[0];
    const float* t  = (const float*)d_in[1];
    const float* W1 = (const float*)d_in[2];
    const float* W2 = (const float*)d_in[3];
    const float* W3 = (const float*)d_in[4];
    const float* b3 = (const float*)d_in[5];
    const float* Wd = (const float*)d_in[6];
    const float* bd = (const float*)d_in[7];
    float* out = (float*)d_out;

    char* ws = (char*)d_ws;
    _Float16* U    = (_Float16*)ws; ws += (size_t)N_B * LQ * CIN * 2;    // 512 KB
    _Float16* T3T  = (_Float16*)ws; ws += (size_t)N_B * COUT * LQ * 2;   //   1 MB
    _Float16* W1t  = (_Float16*)ws; ws += (size_t)DD * CIN * 2;          // 128 KB
    _Float16* W2h  = (_Float16*)ws; ws += (size_t)DD * DD * 2;           // 128 KB
    _Float16* W3h  = (_Float16*)ws; ws += (size_t)COUT * DD * 2;         // 256 KB
    _Float16* Wdh  = (_Float16*)ws; ws += (size_t)COUT * CIN * 2;        // 256 KB
    (void)in_sizes; (void)n_in; (void)out_size; (void)ws_size;

    cvt_f32_to_f16 <<<(DD * DD / 4 + 255) / 256, 256, 0, stream>>>(W2, W2h, DD * DD);
    cvt_f32_to_f16 <<<(COUT * DD / 4 + 255) / 256, 256, 0, stream>>>(W3, W3h, COUT * DD);
    cvt_f32_to_f16 <<<(COUT * CIN / 4 + 255) / 256, 256, 0, stream>>>(Wd, Wdh, COUT * CIN);
    cvt_transpose_f16<<<(DD * CIN + 255) / 256, 256, 0, stream>>>(W1, W1t, DD, CIN);

    prep_kernel<<<N_B, 256, 0, stream>>>(t, b3, W1t, W2h, W3h, U, T3T);
    main_kernel<<<dim3(HW_ / 64, N_B), 256, 0, stream>>>(x, U, T3T, Wdh, bd, out);
}
